// GPT2LinearSelfAttentionOld_53283364274209
// MI455X (gfx1250) — matrix operand order repaired and checked by reading, compiled, not run on hardware
//
#include <hip/hip_runtime.h>
#include <hip/hip_bf16.h>
#include <stdint.h>

// Problem constants (from the reference)
#define BB 8
#define TT 2048
#define CC 1024
#define HH 16
#define DD 64
#define BT (BB*TT)   // 16384 rows
#define N1 (3*CC)    // 3072 qkv cols

typedef __bf16 bf16;
typedef __attribute__((ext_vector_type(16))) __bf16 bf16x16;
typedef __attribute__((ext_vector_type(8)))  float  f32x8;

union Frag32 { uint4 q[2]; bf16x16 v; };
union Half8  { uint4 q;    bf16 h[8]; };

static __device__ __forceinline__ f32x8 wmma_bf16(bf16x16 a, bf16x16 b, f32x8 c) {
  // D = A(16x32 bf16) * B(32x16 bf16) + C(16x16 f32)
  return __builtin_amdgcn_wmma_f32_16x16x32_bf16(false, a, false, b, (short)0, c,
                                                 false, false);
}

// CDNA5 async global->LDS copy (16B per lane), tracked by ASYNCcnt.
// vdst = per-lane LDS byte offset (low 32 bits of generic shared pointer),
// vaddr = 64-bit global address, no SADDR.
static __device__ __forceinline__ void async_g2l_b128(void* lds, const void* g) {
  uint32_t l = (uint32_t)(uintptr_t)lds;
  asm volatile("global_load_async_to_lds_b128 %0, %1, off"
               :: "v"(l), "v"(g) : "memory");
}
static __device__ __forceinline__ void wait_async0() {   // drain all async copies
  asm volatile("s_wait_asynccnt 0x0" ::: "memory");
}
static __device__ __forceinline__ void wait_async6() {   // older 6-op tile done,
  asm volatile("s_wait_asynccnt 0x6" ::: "memory");      // newer 6 still in flight
}

// ---------------------------------------------------------------- converts
__global__ void k_convert(const float* __restrict__ src, bf16* __restrict__ dst, int n) {
  int i = blockIdx.x * blockDim.x + threadIdx.x;
  if (i < n) dst[i] = (bf16)src[i];
}

// dst[c*rows + r] = src[r*cols + c]  (store K-major / transposed weights)
__global__ void k_transpose_convert(const float* __restrict__ src, bf16* __restrict__ dst,
                                    int rows, int cols) {
  int i = blockIdx.x * blockDim.x + threadIdx.x;
  if (i < rows * cols) {
    int r = i / cols, c = i % cols;
    dst[(size_t)c * rows + r] = (bf16)src[i];
  }
}

// ---------------------------------------------------------------- QKV GEMM + epilogue
// X: [BT, CC] bf16 row-major.  WT: [N1, CC] bf16 (w_attn transposed, K contiguous).
// Tile: block = 128(M) x 256(N), 8 waves each 64x64, K-step 32, double-buffered LDS.
__global__ __launch_bounds__(256)
void k_qkv(const bf16* __restrict__ X, const bf16* __restrict__ WT,
           const float* __restrict__ bias, const int* __restrict__ am,
           bf16* __restrict__ phiq, bf16* __restrict__ phikT, bf16* __restrict__ vT)
{
  __shared__ bf16 As[2 * 128 * 40];   // ping-pong, row stride 40 halves
  __shared__ bf16 Bs[2 * 256 * 40];
  const int bM = blockIdx.x, bN = blockIdx.y;
  const int tid = threadIdx.x, lane = tid & 31, wave = tid >> 5;
  const int wm = wave >> 2, wn = wave & 3;       // 2 x 4 wave grid
  const int sel = lane >> 4, ln = lane & 15;
  const int r = tid >> 1, ko = (tid & 1) * 16;

  auto stage = [&](int k0, int buf) {  // 6 async b128 per thread -> 24KB tile pair
    bf16* as = As + buf * (128 * 40);
    bf16* bs = Bs + buf * (256 * 40);
    const bf16* ga = X + (size_t)(bM * 128 + r) * CC + k0 + ko;
    async_g2l_b128(as + r * 40 + ko,     ga);
    async_g2l_b128(as + r * 40 + ko + 8, ga + 8);
    const bf16* gb = WT + (size_t)(bN * 256 + tid) * CC + k0;
    async_g2l_b128(bs + tid * 40 + 0,  gb);
    async_g2l_b128(bs + tid * 40 + 8,  gb + 8);
    async_g2l_b128(bs + tid * 40 + 16, gb + 16);
    async_g2l_b128(bs + tid * 40 + 24, gb + 24);
  };

  f32x8 acc[4][4] = {};
  stage(0, 0);

  for (int k0 = 0, buf = 0; k0 < CC; k0 += 32, buf ^= 1) {
    const bool more = (k0 + 32) < CC;
    if (more) stage(k0 + 32, buf ^ 1);   // overlap next tile with this one's compute
    if (more) wait_async6(); else wait_async0();
    __syncthreads();

    const bf16* as = As + buf * (128 * 40);
    const bf16* bs = Bs + buf * (256 * 40);
    bf16x16 afr[4];
#pragma unroll
    for (int mi = 0; mi < 4; ++mi) {   // A frag: lane holds row M=ln, K-half per sel
      int row = wm * 64 + mi * 16 + ln;
      Frag32 f;
      f.q[0] = *(const uint4*)(as + row * 40 + sel * 8);        // K 0-7 / 8-15
      f.q[1] = *(const uint4*)(as + row * 40 + 16 + sel * 8);   // K 16-23 / 24-31
      afr[mi] = f.v;
    }
#pragma unroll
    for (int ni = 0; ni < 4; ++ni) {   // B frag: lane holds col N=ln, K-half per sel
      int n = wn * 64 + ni * 16 + ln;
      Frag32 f;
      f.q[0] = *(const uint4*)(bs + n * 40 + sel * 8);
      f.q[1] = *(const uint4*)(bs + n * 40 + 16 + sel * 8);
      bf16x16 bfr = f.v;
#pragma unroll
      for (int mi = 0; mi < 4; ++mi)
        acc[mi][ni] = wmma_bf16(afr[mi], bfr, acc[mi][ni]);
    }
    __syncthreads();   // all waves done reading `buf` before it is re-staged
  }

  // fused epilogue: bias + elu(+1) + mask, scatter into attention layouts
#pragma unroll
  for (int mi = 0; mi < 4; ++mi)
#pragma unroll
    for (int ni = 0; ni < 4; ++ni)
      for (int rr = 0; rr < 8; ++rr) {
        int M  = bM * 128 + wm * 64 + mi * 16 + rr + sel * 8;  // C/D layout: M=r / r+8
        int Nj = bN * 256 + wn * 64 + ni * 16 + ln;
        int b = M / TT, t = M % TT;
        int sec = Nj / CC, c = Nj % CC, h = c / DD, d = c % DD;
        float val = acc[mi][ni][rr] + bias[Nj];
        float m = (am[b * TT + t] > 0) ? 1.f : 0.f;
        size_t bh = (size_t)(b * HH + h);
        if (sec == 0) {                       // phi_q = elu(q)+1, [B,H,T,D]
          float p = val > 0.f ? val + 1.f : __expf(val);
          phiq[(bh * TT + t) * DD + d] = (bf16)p;
        } else if (sec == 1) {                // phi_k^T * mask, [B,H,D,T]
          float p = (val > 0.f ? val + 1.f : __expf(val)) * m;
          phikT[(bh * DD + d) * TT + t] = (bf16)p;
        } else {                              // v^T * mask, [B,H,D,T]
          vT[(bh * DD + d) * TT + t] = (bf16)(val * m);
        }
      }
}

// ---------------------------------------------------------------- kv = phi_k^T @ v, k_sum
// One wave per (b,h): 64x64 output, K = T = 2048. Both operands K-contiguous.
__global__ __launch_bounds__(32)
void k_kv(const bf16* __restrict__ phikT, const bf16* __restrict__ vT,
          bf16* __restrict__ kvT, float* __restrict__ ksum)
{
  const int bh = blockIdx.x;
  const int lane = threadIdx.x & 31, sel = lane >> 4, ln = lane & 15;
  const bf16* A  = phikT + (size_t)bh * DD * TT;   // [d, t]
  const bf16* Bt = vT    + (size_t)bh * DD * TT;   // [f, t]
  f32x8 acc[4][4] = {};

  for (int k0 = 0; k0 < TT; k0 += 32) {
    bf16x16 afr[4], bfr[4];
#pragma unroll
    for (int mi = 0; mi < 4; ++mi) {
      const bf16* p = A + (size_t)(mi * 16 + ln) * TT + k0;
      Frag32 f;
      f.q[0] = *(const uint4*)(p + sel * 8);
      f.q[1] = *(const uint4*)(p + 16 + sel * 8);
      afr[mi] = f.v;
    }
#pragma unroll
    for (int ni = 0; ni < 4; ++ni) {
      const bf16* p = Bt + (size_t)(ni * 16 + ln) * TT + k0;
      Frag32 f;
      f.q[0] = *(const uint4*)(p + sel * 8);
      f.q[1] = *(const uint4*)(p + 16 + sel * 8);
      bfr[ni] = f.v;
    }
#pragma unroll
    for (int mi = 0; mi < 4; ++mi)
#pragma unroll
      for (int ni = 0; ni < 4; ++ni)
        acc[mi][ni] = wmma_bf16(afr[mi], bfr[ni], acc[mi][ni]);
  }

  // store kv transposed ([f, d]) in bf16 for the numerator GEMM
#pragma unroll
  for (int mi = 0; mi < 4; ++mi)
#pragma unroll
    for (int ni = 0; ni < 4; ++ni)
      for (int rr = 0; rr < 8; ++rr) {
        int d = mi * 16 + rr + sel * 8;
        int f = ni * 16 + ln;
        kvT[(size_t)bh * DD * DD + (size_t)f * DD + d] = (bf16)acc[mi][ni][rr];
      }

  // k_sum[d] = sum_t phi_k[t,d]; 2 rows per lane, vectorized 8-wide
  for (int j = 0; j < 2; ++j) {
    int d = lane * 2 + j;
    const bf16* row = A + (size_t)d * TT;
    float s = 0.f;
    for (int t = 0; t < TT; t += 8) {
      Half8 v8;
      v8.q = *(const uint4*)(row + t);
#pragma unroll
      for (int e = 0; e < 8; ++e) s += (float)v8.h[e];
    }
    ksum[bh * DD + d] = s;
  }
}

// ---------------------------------------------------------------- y = (phi_q @ kv) / den
// Block: 8 waves, each a 16(t) x 64(f) tile of one (b,h); K = D = 64.
__global__ __launch_bounds__(256)
void k_attn(const bf16* __restrict__ phiq, const bf16* __restrict__ kvT,
            const float* __restrict__ ksum, bf16* __restrict__ Y)
{
  __shared__ float den_s[8][16];
  const int tb = blockIdx.x, bh = blockIdx.y;
  const int tid = threadIdx.x, lane = tid & 31, wave = tid >> 5;
  const int sel = lane >> 4, ln = lane & 15;
  const int t0 = tb * 128 + wave * 16;
  const bf16* A  = phiq + (size_t)bh * TT * DD;   // [t, d]
  const bf16* Bt = kvT  + (size_t)bh * DD * DD;   // [f, d]

  if (lane < 16) {  // denominator for the 16 rows of this wave's tile
    const bf16* row = A + (size_t)(t0 + lane) * DD;
    float s = 0.f;
    for (int d = 0; d < DD; ++d) s += (float)row[d] * ksum[bh * DD + d];
    den_s[wave][lane] = fmaxf(s, 1e-8f);
  }
  __syncthreads();

  f32x8 acc[4] = {};
#pragma unroll
  for (int k0 = 0; k0 < DD; k0 += 32) {
    const bf16* pa = A + (size_t)(t0 + ln) * DD + k0;
    Frag32 fa;
    fa.q[0] = *(const uint4*)(pa + sel * 8);
    fa.q[1] = *(const uint4*)(pa + 16 + sel * 8);
    bf16x16 afr = fa.v;
#pragma unroll
    for (int ni = 0; ni < 4; ++ni) {
      const bf16* pb = Bt + (size_t)(ni * 16 + ln) * DD + k0;
      Frag32 fb;
      fb.q[0] = *(const uint4*)(pb + sel * 8);
      fb.q[1] = *(const uint4*)(pb + 16 + sel * 8);
      acc[ni] = wmma_bf16(afr, fb.v, acc[ni]);
    }
  }

  int b = bh / HH, h = bh % HH;
#pragma unroll
  for (int ni = 0; ni < 4; ++ni)
    for (int rr = 0; rr < 8; ++rr) {
      int t = t0 + rr + sel * 8;
      int f = ni * 16 + ln;
      float y = acc[ni][rr] / den_s[wave][rr + sel * 8];
      Y[((size_t)b * TT + t) * CC + h * DD + f] = (bf16)y;   // [B,T,C] directly
    }
}

// ---------------------------------------------------------------- out = Y @ w_proj + b
__global__ __launch_bounds__(256)
void k_proj(const bf16* __restrict__ Yb, const bf16* __restrict__ WT,
            const float* __restrict__ bias, float* __restrict__ out)
{
  __shared__ bf16 As[2 * 128 * 40];
  __shared__ bf16 Bs[2 * 256 * 40];
  const int bM = blockIdx.x, bN = blockIdx.y;
  const int tid = threadIdx.x, lane = tid & 31, wave = tid >> 5;
  const int wm = wave >> 2, wn = wave & 3;
  const int sel = lane >> 4, ln = lane & 15;
  const int r = tid >> 1, ko = (tid & 1) * 16;

  auto stage = [&](int k0, int buf) {
    bf16* as = As + buf * (128 * 40);
    bf16* bs = Bs + buf * (256 * 40);
    const bf16* ga = Yb + (size_t)(bM * 128 + r) * CC + k0 + ko;
    async_g2l_b128(as + r * 40 + ko,     ga);
    async_g2l_b128(as + r * 40 + ko + 8, ga + 8);
    const bf16* gb = WT + (size_t)(bN * 256 + tid) * CC + k0;
    async_g2l_b128(bs + tid * 40 + 0,  gb);
    async_g2l_b128(bs + tid * 40 + 8,  gb + 8);
    async_g2l_b128(bs + tid * 40 + 16, gb + 16);
    async_g2l_b128(bs + tid * 40 + 24, gb + 24);
  };

  f32x8 acc[4][4] = {};
  stage(0, 0);

  for (int k0 = 0, buf = 0; k0 < CC; k0 += 32, buf ^= 1) {
    const bool more = (k0 + 32) < CC;
    if (more) stage(k0 + 32, buf ^ 1);
    if (more) wait_async6(); else wait_async0();
    __syncthreads();

    const bf16* as = As + buf * (128 * 40);
    const bf16* bs = Bs + buf * (256 * 40);
    bf16x16 afr[4];
#pragma unroll
    for (int mi = 0; mi < 4; ++mi) {
      int row = wm * 64 + mi * 16 + ln;
      Frag32 f;
      f.q[0] = *(const uint4*)(as + row * 40 + sel * 8);
      f.q[1] = *(const uint4*)(as + row * 40 + 16 + sel * 8);
      afr[mi] = f.v;
    }
#pragma unroll
    for (int ni = 0; ni < 4; ++ni) {
      int n = wn * 64 + ni * 16 + ln;
      Frag32 f;
      f.q[0] = *(const uint4*)(bs + n * 40 + sel * 8);
      f.q[1] = *(const uint4*)(bs + n * 40 + 16 + sel * 8);
      bf16x16 bfr = f.v;
#pragma unroll
      for (int mi = 0; mi < 4; ++mi)
        acc[mi][ni] = wmma_bf16(afr[mi], bfr, acc[mi][ni]);
    }
    __syncthreads();
  }

#pragma unroll
  for (int mi = 0; mi < 4; ++mi)
#pragma unroll
    for (int ni = 0; ni < 4; ++ni)
      for (int rr = 0; rr < 8; ++rr) {
        int M  = bM * 128 + wm * 64 + mi * 16 + rr + sel * 8;
        int Nj = bN * 256 + wn * 64 + ni * 16 + ln;
        out[(size_t)M * CC + Nj] = acc[mi][ni][rr] + bias[Nj];
      }
}

// ---------------------------------------------------------------- launcher
extern "C" void kernel_launch(void* const* d_in, const int* in_sizes, int n_in,
                              void* d_out, int out_size, void* d_ws, size_t ws_size,
                              hipStream_t stream) {
  (void)in_sizes; (void)n_in; (void)out_size; (void)ws_size;
  const float* x      = (const float*)d_in[0];
  const int*   am     = (const int*)d_in[1];
  const float* w_attn = (const float*)d_in[2];
  const float* b_attn = (const float*)d_in[3];
  const float* w_proj = (const float*)d_in[4];
  const float* b_proj = (const float*)d_in[5];
  float* out = (float*)d_out;

  char* ws = (char*)d_ws;
  size_t o = 0;
  auto alloc = [&](size_t bytes) -> void* {
    void* p = ws + o;
    o = (o + bytes + 255) & ~(size_t)255;
    return p;
  };
  bf16*  xb     = (bf16*)alloc((size_t)BT * CC * 2);
  bf16*  wattnT = (bf16*)alloc((size_t)N1 * CC * 2);
  bf16*  wprojT = (bf16*)alloc((size_t)CC * CC * 2);
  bf16*  phiq   = (bf16*)alloc((size_t)BB * HH * TT * DD * 2);
  bf16*  phikT  = (bf16*)alloc((size_t)BB * HH * TT * DD * 2);
  bf16*  vT     = (bf16*)alloc((size_t)BB * HH * TT * DD * 2);
  bf16*  kvT    = (bf16*)alloc((size_t)BB * HH * DD * DD * 2);
  float* ksum   = (float*)alloc((size_t)BB * HH * DD * 4);
  bf16*  Yb     = (bf16*)alloc((size_t)BT * CC * 2);

  k_convert<<<(BT * CC + 255) / 256, 256, 0, stream>>>(x, xb, BT * CC);
  k_transpose_convert<<<(CC * N1 + 255) / 256, 256, 0, stream>>>(w_attn, wattnT, CC, N1);
  k_transpose_convert<<<(CC * CC + 255) / 256, 256, 0, stream>>>(w_proj, wprojT, CC, CC);

  k_qkv<<<dim3(BT / 128, N1 / 256), 256, 0, stream>>>(xb, wattnT, b_attn, am,
                                                      phiq, phikT, vT);
  k_kv<<<BB * HH, 32, 0, stream>>>(phikT, vT, kvT, ksum);
  k_attn<<<dim3(TT / 128, BB * HH), 256, 0, stream>>>(phiq, kvT, ksum, Yb);
  k_proj<<<dim3(BT / 128, CC / 256), 256, 0, stream>>>(Yb, wprojT, b_proj, out);
}
